// RETROModelBlock_87823491269303
// MI455X (gfx1250) — compile-verified, matrix operand order checked by reading, not run on hardware
//
#include <hip/hip_runtime.h>
#include <hip/hip_bf16.h>
#include <cstddef>

// ---------------------------------------------------------------------------
// Types
// ---------------------------------------------------------------------------
typedef __bf16 bf16_t;
typedef __bf16  bf16x16 __attribute__((ext_vector_type(16)));
typedef float   floatx8 __attribute__((ext_vector_type(8)));
typedef unsigned int u32x4 __attribute__((ext_vector_type(4)));

#define HEADS 16
#define DIM_HEAD 64
#define DMODEL 1024
#define NTOK 1024
#define BATCH 4
#define ROWS (BATCH * NTOK)          // 4096 token rows
#define NCHUNK 64                    // b*kc
#define KVLEN 256                    // r*rn per chunk (w/o null)
#define KROWS (NCHUNK * KVLEN)       // 16384 ctx rows
#define DFF 4096

// ---------------------------------------------------------------------------
// WMMA fragment load from LDS.
// Tile stored row-major [16 rows][ldk bf16], row = lane&15, kbase = 8*(lane>>4),
// elements 0..7 -> k = kbase..kbase+7, elements 8..15 -> k = kbase+16..kbase+23.
// (Matches CDNA5 ISA 16-bit A 16x32 layout; B uses the mirrored layout with
//  the tile stored as [n][k], i.e. column-major relative to memory B[K,N].)
// ---------------------------------------------------------------------------
__device__ __forceinline__ bf16x16 frag_ld(const bf16_t* base, int ldk) {
    int l = threadIdx.x & 31;
    const bf16_t* p = base + (l & 15) * ldk + ((l >> 4) << 3);
    union { bf16x16 v; u32x4 q[2]; } u;
    u.q[0] = *reinterpret_cast<const u32x4*>(p);
    u.q[1] = *reinterpret_cast<const u32x4*>(p + 16);
    return u.v;
}

__device__ __forceinline__ floatx8 wmma_bf16(bf16x16 a, bf16x16 b, floatx8 c) {
    return __builtin_amdgcn_wmma_f32_16x16x32_bf16(false, a, false, b, (short)0, c,
                                                   false, false);
}

__device__ __forceinline__ float gelu_tanh(float x) {
    const float c = 0.7978845608028654f;  // sqrt(2/pi)
    return 0.5f * x * (1.0f + tanhf(c * (x + 0.044715f * x * x * x)));
}

// ---------------------------------------------------------------------------
// Generic bf16 WMMA GEMM: C[M,N](f32 acc) = alpha * A[M,K] x B[K,N] + bias
// A, B bf16 row-major. Optional fp32 and/or bf16 outputs, optional GELU.
// Block tile 128x128, K-step 32, 256 threads (8 waves), wave tile 32x64.
// ---------------------------------------------------------------------------
__global__ __launch_bounds__(256) void gemm_bf16_kernel(
    const bf16_t* __restrict__ A, const bf16_t* __restrict__ B,
    const float* __restrict__ bias, float* __restrict__ Cf,
    bf16_t* __restrict__ Cb, int M, int N, int K, float alpha, int act_gelu) {
    __shared__ alignas(16) bf16_t As[128 * 32];   // [m][k]
    __shared__ alignas(16) bf16_t Bs[128 * 32];   // [n][k] (transposed on store)

    const int tid = threadIdx.x;
    const int lane = tid & 31, wave = tid >> 5;
    const int m0 = blockIdx.y * 128, n0 = blockIdx.x * 128;
    const int wm = (wave >> 1) * 32;   // wave row offset within block tile
    const int wn = (wave & 1) * 64;    // wave col offset within block tile

    floatx8 acc[2][4];
#pragma unroll
    for (int i = 0; i < 2; ++i)
#pragma unroll
        for (int j = 0; j < 4; ++j)
#pragma unroll
            for (int e = 0; e < 8; ++e) acc[i][j][e] = 0.0f;

    const int arow = tid >> 1, acol = (tid & 1) << 4;   // A: 16 bf16 per thread
    const int bk = tid & 31, bn = (tid >> 5) << 4;      // B: 16 bf16 per thread

    for (int k0 = 0; k0 < K; k0 += 32) {
        __syncthreads();
        // A tile 128x32 -> LDS row-major
        {
            const u32x4* s =
                reinterpret_cast<const u32x4*>(A + (size_t)(m0 + arow) * K + k0 + acol);
            u32x4* d = reinterpret_cast<u32x4*>(As + arow * 32 + acol);
            d[0] = s[0];
            d[1] = s[1];
        }
        // B tile 32x128 -> LDS transposed [n][k]
        {
            union { u32x4 q[2]; bf16_t h[16]; } u;
            const u32x4* s =
                reinterpret_cast<const u32x4*>(B + (size_t)(k0 + bk) * N + n0 + bn);
            u.q[0] = s[0];
            u.q[1] = s[1];
#pragma unroll
            for (int i = 0; i < 16; ++i) Bs[(bn + i) * 32 + bk] = u.h[i];
        }
        if (k0 + 32 < K) {
            __builtin_prefetch(A + (size_t)(m0 + arow) * K + (k0 + 32) + acol, 0, 1);
            __builtin_prefetch(B + (size_t)(k0 + 32 + bk) * N + n0 + bn, 0, 1);
        }
        __syncthreads();

        bf16x16 af[2], bfr[4];
#pragma unroll
        for (int i = 0; i < 2; ++i) af[i] = frag_ld(As + (wm + i * 16) * 32, 32);
#pragma unroll
        for (int j = 0; j < 4; ++j) bfr[j] = frag_ld(Bs + (wn + j * 16) * 32, 32);
#pragma unroll
        for (int i = 0; i < 2; ++i)
#pragma unroll
            for (int j = 0; j < 4; ++j) acc[i][j] = wmma_bf16(af[i], bfr[j], acc[i][j]);
    }

    const int half = lane >> 4, ln = lane & 15;
#pragma unroll
    for (int i = 0; i < 2; ++i)
#pragma unroll
        for (int j = 0; j < 4; ++j)
#pragma unroll
            for (int e = 0; e < 8; ++e) {
                int m = m0 + wm + i * 16 + e + half * 8;
                int n = n0 + wn + j * 16 + ln;
                float v = acc[i][j][e] * alpha + (bias ? bias[n] : 0.0f);
                if (act_gelu) v = gelu_tanh(v);
                size_t idx = (size_t)m * N + n;
                if (Cf) Cf[idx] = v;
                if (Cb) Cb[idx] = (bf16_t)v;
            }
}

// ---------------------------------------------------------------------------
// Flash self-attention (causal). q/k/v/out: [4096, 1024] bf16, merged-head
// layout (row = b*1024+t, col = h*64+d). Grid (16 q-tiles, 64 b*h), 128 thr.
// ---------------------------------------------------------------------------
__global__ __launch_bounds__(128) void attn_self_kernel(
    const bf16_t* __restrict__ q, const bf16_t* __restrict__ k,
    const bf16_t* __restrict__ v, bf16_t* __restrict__ out) {
    __shared__ alignas(16) bf16_t Qs[64 * 64];
    __shared__ alignas(16) bf16_t Ks[64 * 64];
    __shared__ alignas(16) bf16_t Vt[64 * 64];  // [dim][kv]
    __shared__ alignas(16) bf16_t Ps[64 * 64];

    const int qt = blockIdx.x;              // 0..15
    const int bh = blockIdx.y;              // 0..63
    const int b = bh >> 4, h = bh & 15;
    const int tid = threadIdx.x, lane = tid & 31, wave = tid >> 5;
    const int half = lane >> 4, ln = lane & 15;
    const size_t rowBase = (size_t)b * NTOK;

    {   // Q tile
        int r = tid >> 1, c0 = (tid & 1) * 32;
        const u32x4* s = reinterpret_cast<const u32x4*>(
            q + (rowBase + qt * 64 + r) * DMODEL + h * 64 + c0);
        u32x4* d = reinterpret_cast<u32x4*>(Qs + r * 64 + c0);
        d[0] = s[0]; d[1] = s[1]; d[2] = s[2]; d[3] = s[3];
    }

    float m_run[8], l_run[8];
    floatx8 Oacc[4];
#pragma unroll
    for (int e = 0; e < 8; ++e) { m_run[e] = -1e30f; l_run[e] = 0.0f; }
#pragma unroll
    for (int nt = 0; nt < 4; ++nt)
#pragma unroll
        for (int e = 0; e < 8; ++e) Oacc[nt][e] = 0.0f;

    for (int jt = 0; jt <= qt; ++jt) {
        __syncthreads();
        {   // K tile row-major
            int r = tid >> 1, c0 = (tid & 1) * 32;
            const u32x4* s = reinterpret_cast<const u32x4*>(
                k + (rowBase + jt * 64 + r) * DMODEL + h * 64 + c0);
            u32x4* d = reinterpret_cast<u32x4*>(Ks + r * 64 + c0);
            d[0] = s[0]; d[1] = s[1]; d[2] = s[2]; d[3] = s[3];
        }
        {   // V tile transposed [dim][kv]
            int r = tid & 63, dg = tid >> 6;  // dg 0..1 -> dims dg*32..+32
            union { u32x4 q4[4]; bf16_t hh[32]; } u;
            const u32x4* s = reinterpret_cast<const u32x4*>(
                v + (rowBase + jt * 64 + r) * DMODEL + h * 64 + dg * 32);
            u.q4[0] = s[0]; u.q4[1] = s[1]; u.q4[2] = s[2]; u.q4[3] = s[3];
#pragma unroll
            for (int i = 0; i < 32; ++i) Vt[(dg * 32 + i) * 64 + r] = u.hh[i];
        }
        __syncthreads();

        floatx8 sacc[4];
#pragma unroll
        for (int ct = 0; ct < 4; ++ct)
#pragma unroll
            for (int e = 0; e < 8; ++e) sacc[ct][e] = 0.0f;
#pragma unroll
        for (int k0 = 0; k0 < 64; k0 += 32) {
            bf16x16 a = frag_ld(Qs + (wave * 16) * 64 + k0, 64);
#pragma unroll
            for (int ct = 0; ct < 4; ++ct) {
                bf16x16 bb = frag_ld(Ks + (ct * 16) * 64 + k0, 64);
                sacc[ct] = wmma_bf16(a, bb, sacc[ct]);
            }
        }
        if (jt == qt) {  // causal mask on diagonal tile
#pragma unroll
            for (int ct = 0; ct < 4; ++ct)
#pragma unroll
                for (int e = 0; e < 8; ++e) {
                    int row = wave * 16 + e + half * 8;
                    int col = ct * 16 + ln;
                    if (col > row) sacc[ct][e] = -1e30f;
                }
        }
        // online softmax
        float mt[8], p[4][8], rs[8];
#pragma unroll
        for (int e = 0; e < 8; ++e) {
            mt[e] = sacc[0][e];
#pragma unroll
            for (int ct = 1; ct < 4; ++ct) mt[e] = fmaxf(mt[e], sacc[ct][e]);
#pragma unroll
            for (int xm = 1; xm < 16; xm <<= 1)
                mt[e] = fmaxf(mt[e], __shfl_xor(mt[e], xm, 32));
        }
#pragma unroll
        for (int e = 0; e < 8; ++e) {
            float mn = fmaxf(m_run[e], mt[e]);
            float sc = __expf(m_run[e] - mn);
            rs[e] = 0.0f;
#pragma unroll
            for (int ct = 0; ct < 4; ++ct) {
                p[ct][e] = __expf(sacc[ct][e] - mn);
                rs[e] += p[ct][e];
            }
#pragma unroll
            for (int xm = 1; xm < 16; xm <<= 1) rs[e] += __shfl_xor(rs[e], xm, 32);
            l_run[e] = l_run[e] * sc + rs[e];
            m_run[e] = mn;
#pragma unroll
            for (int nt = 0; nt < 4; ++nt) Oacc[nt][e] *= sc;
        }
        // P -> LDS (C-layout -> A-layout re-tiling), per-wave private region
#pragma unroll
        for (int ct = 0; ct < 4; ++ct)
#pragma unroll
            for (int e = 0; e < 8; ++e)
                Ps[(wave * 16 + e + half * 8) * 64 + ct * 16 + ln] = (bf16_t)p[ct][e];
        // O += P x V
#pragma unroll
        for (int kv0 = 0; kv0 < 64; kv0 += 32) {
            bf16x16 a = frag_ld(Ps + (wave * 16) * 64 + kv0, 64);
#pragma unroll
            for (int nt = 0; nt < 4; ++nt) {
                bf16x16 bb = frag_ld(Vt + (nt * 16) * 64 + kv0, 64);
                Oacc[nt] = wmma_bf16(a, bb, Oacc[nt]);
            }
        }
    }
    // write O (merged heads)
#pragma unroll
    for (int nt = 0; nt < 4; ++nt)
#pragma unroll
        for (int e = 0; e < 8; ++e) {
            int row = qt * 64 + wave * 16 + e + half * 8;
            out[(rowBase + row) * DMODEL + h * 64 + nt * 16 + ln] =
                (bf16_t)(Oacc[nt][e] / l_run[e]);
        }
}

// ---------------------------------------------------------------------------
// Flash cross-attention with null kv folded into the softmax initial state.
// q/out: [4096, 1024] bf16 (row = chunk*64 + r). k/v: [16384, 1024] bf16
// (row = chunk*256 + kv). Grid (64 chunks, 16 heads), 128 threads.
// ---------------------------------------------------------------------------
__global__ __launch_bounds__(128) void attn_cross_kernel(
    const bf16_t* __restrict__ q, const bf16_t* __restrict__ k,
    const bf16_t* __restrict__ v, const float* __restrict__ null_k,
    const float* __restrict__ null_v, bf16_t* __restrict__ out) {
    __shared__ alignas(16) bf16_t Qs[64 * 64];
    __shared__ alignas(16) bf16_t Ks[64 * 64];
    __shared__ alignas(16) bf16_t Vt[64 * 64];
    __shared__ alignas(16) bf16_t Ps[64 * 64];
    __shared__ float sn[64];

    const int chunk = blockIdx.x;   // 0..63
    const int h = blockIdx.y;       // 0..15
    const int tid = threadIdx.x, lane = tid & 31, wave = tid >> 5;
    const int half = lane >> 4, ln = lane & 15;

    {   // Q tile
        int r = tid >> 1, c0 = (tid & 1) * 32;
        const u32x4* s = reinterpret_cast<const u32x4*>(
            q + ((size_t)chunk * 64 + r) * DMODEL + h * 64 + c0);
        u32x4* d = reinterpret_cast<u32x4*>(Qs + r * 64 + c0);
        d[0] = s[0]; d[1] = s[1]; d[2] = s[2]; d[3] = s[3];
    }
    __syncthreads();
    if (tid < 64) {  // s_null per q row
        float acc = 0.0f;
        for (int i = 0; i < 64; ++i)
            acc += (float)Qs[tid * 64 + i] * null_k[h * 64 + i];
        sn[tid] = acc;
    }
    __syncthreads();

    float m_run[8], l_run[8];
    floatx8 Oacc[4];
#pragma unroll
    for (int e = 0; e < 8; ++e) {
        m_run[e] = sn[wave * 16 + e + half * 8];
        l_run[e] = 1.0f;
    }
#pragma unroll
    for (int nt = 0; nt < 4; ++nt) {
        float nv = null_v[h * 64 + nt * 16 + ln];
#pragma unroll
        for (int e = 0; e < 8; ++e) Oacc[nt][e] = nv;
    }

    for (int jt = 0; jt < 4; ++jt) {
        __syncthreads();
        {   // K tile
            int r = tid >> 1, c0 = (tid & 1) * 32;
            const u32x4* s = reinterpret_cast<const u32x4*>(
                k + ((size_t)chunk * KVLEN + jt * 64 + r) * DMODEL + h * 64 + c0);
            u32x4* d = reinterpret_cast<u32x4*>(Ks + r * 64 + c0);
            d[0] = s[0]; d[1] = s[1]; d[2] = s[2]; d[3] = s[3];
        }
        {   // V tile transposed
            int r = tid & 63, dg = tid >> 6;
            union { u32x4 q4[4]; bf16_t hh[32]; } u;
            const u32x4* s = reinterpret_cast<const u32x4*>(
                v + ((size_t)chunk * KVLEN + jt * 64 + r) * DMODEL + h * 64 + dg * 32);
            u.q4[0] = s[0]; u.q4[1] = s[1]; u.q4[2] = s[2]; u.q4[3] = s[3];
#pragma unroll
            for (int i = 0; i < 32; ++i) Vt[(dg * 32 + i) * 64 + r] = u.hh[i];
        }
        __syncthreads();

        floatx8 sacc[4];
#pragma unroll
        for (int ct = 0; ct < 4; ++ct)
#pragma unroll
            for (int e = 0; e < 8; ++e) sacc[ct][e] = 0.0f;
#pragma unroll
        for (int k0 = 0; k0 < 64; k0 += 32) {
            bf16x16 a = frag_ld(Qs + (wave * 16) * 64 + k0, 64);
#pragma unroll
            for (int ct = 0; ct < 4; ++ct) {
                bf16x16 bb = frag_ld(Ks + (ct * 16) * 64 + k0, 64);
                sacc[ct] = wmma_bf16(a, bb, sacc[ct]);
            }
        }
        float mt[8], p[4][8], rs[8];
#pragma unroll
        for (int e = 0; e < 8; ++e) {
            mt[e] = sacc[0][e];
#pragma unroll
            for (int ct = 1; ct < 4; ++ct) mt[e] = fmaxf(mt[e], sacc[ct][e]);
#pragma unroll
            for (int xm = 1; xm < 16; xm <<= 1)
                mt[e] = fmaxf(mt[e], __shfl_xor(mt[e], xm, 32));
        }
#pragma unroll
        for (int e = 0; e < 8; ++e) {
            float mn = fmaxf(m_run[e], mt[e]);
            float sc = __expf(m_run[e] - mn);
            rs[e] = 0.0f;
#pragma unroll
            for (int ct = 0; ct < 4; ++ct) {
                p[ct][e] = __expf(sacc[ct][e] - mn);
                rs[e] += p[ct][e];
            }
#pragma unroll
            for (int xm = 1; xm < 16; xm <<= 1) rs[e] += __shfl_xor(rs[e], xm, 32);
            l_run[e] = l_run[e] * sc + rs[e];
            m_run[e] = mn;
#pragma unroll
            for (int nt = 0; nt < 4; ++nt) Oacc[nt][e] *= sc;
        }
#pragma unroll
        for (int ct = 0; ct < 4; ++ct)
#pragma unroll
            for (int e = 0; e < 8; ++e)
                Ps[(wave * 16 + e + half * 8) * 64 + ct * 16 + ln] = (bf16_t)p[ct][e];
#pragma unroll
        for (int kv0 = 0; kv0 < 64; kv0 += 32) {
            bf16x16 a = frag_ld(Ps + (wave * 16) * 64 + kv0, 64);
#pragma unroll
            for (int nt = 0; nt < 4; ++nt) {
                bf16x16 bb = frag_ld(Vt + (nt * 16) * 64 + kv0, 64);
                Oacc[nt] = wmma_bf16(a, bb, Oacc[nt]);
            }
        }
    }
#pragma unroll
    for (int nt = 0; nt < 4; ++nt)
#pragma unroll
        for (int e = 0; e < 8; ++e) {
            int row = chunk * 64 + wave * 16 + e + half * 8;
            out[(size_t)row * DMODEL + h * 64 + nt * 16 + ln] =
                (bf16_t)(Oacc[nt][e] / l_run[e]);
        }
}

// ---------------------------------------------------------------------------
// LayerNorm over d=1024 (optionally adds a residual first). Grid = rows.
// ---------------------------------------------------------------------------
__global__ __launch_bounds__(256) void ln_kernel(
    const float* __restrict__ x, const float* __restrict__ res,
    const float* __restrict__ g, const float* __restrict__ b,
    float* __restrict__ yf, bf16_t* __restrict__ yb) {
    __shared__ float red[256];
    const int row = blockIdx.x, tid = threadIdx.x;
    const float* xr = x + (size_t)row * DMODEL;
    const float* rr = res ? res + (size_t)row * DMODEL : nullptr;
    float loc[4], s = 0.0f;
#pragma unroll
    for (int i = 0; i < 4; ++i) {
        int c = tid + i * 256;
        float v = xr[c];
        if (rr) v += rr[c];
        loc[i] = v;
        s += v;
    }
    red[tid] = s;
    __syncthreads();
    for (int st = 128; st > 0; st >>= 1) {
        if (tid < st) red[tid] += red[tid + st];
        __syncthreads();
    }
    float mu = red[0] * (1.0f / 1024.0f);
    __syncthreads();
    float s2 = 0.0f;
#pragma unroll
    for (int i = 0; i < 4; ++i) {
        float d = loc[i] - mu;
        s2 += d * d;
    }
    red[tid] = s2;
    __syncthreads();
    for (int st = 128; st > 0; st >>= 1) {
        if (tid < st) red[tid] += red[tid + st];
        __syncthreads();
    }
    float inv = rsqrtf(red[0] * (1.0f / 1024.0f) + 1e-5f);
#pragma unroll
    for (int i = 0; i < 4; ++i) {
        int c = tid + i * 256;
        float yv = (loc[i] - mu) * inv * g[c] + b[c];
        if (yf) yf[(size_t)row * DMODEL + c] = yv;
        if (yb) yb[(size_t)row * DMODEL + c] = (bf16_t)yv;
    }
}

// ---------------------------------------------------------------------------
// cca shift (pad 63 front, truncate) + residual add + T5 norm -> bf16
// ---------------------------------------------------------------------------
__global__ __launch_bounds__(256) void t5norm_kernel(
    const float* __restrict__ hs, const float* __restrict__ cca,
    const float* __restrict__ w, bf16_t* __restrict__ yb) {
    __shared__ float red[256];
    const int row = blockIdx.x, tid = threadIdx.x;  // row = b*1024 + p
    const int p = row & 1023, b = row >> 10;
    const float* hr = hs + (size_t)row * DMODEL;
    const float* cr =
        (p >= 63) ? cca + ((size_t)(b * NTOK + p - 63)) * DMODEL : nullptr;
    float loc[4], s2 = 0.0f;
#pragma unroll
    for (int i = 0; i < 4; ++i) {
        int c = tid + i * 256;
        float v = hr[c] + (cr ? cr[c] : 0.0f);
        loc[i] = v;
        s2 += v * v;
    }
    red[tid] = s2;
    __syncthreads();
    for (int st = 128; st > 0; st >>= 1) {
        if (tid < st) red[tid] += red[tid + st];
        __syncthreads();
    }
    float norm = sqrtf(red[0]) * (1.0f / 32.0f);  // * d^-0.5
    float inv = 1.0f / fmaxf(norm, 1e-6f);
#pragma unroll
    for (int i = 0; i < 4; ++i) {
        int c = tid + i * 256;
        yb[(size_t)row * DMODEL + c] = (bf16_t)(loc[i] * inv * w[c]);
    }
}

// ---------------------------------------------------------------------------
// Rotary (full head rot=64, half=32). mode 0: q (freq only at pos 0, from
// q_pos_emb row 63 -- the reference's pad of q_pos_emb[:, :, 63:]).
// mode 1: k (freq = k_pos_emb[pos % 128]). fp32 in -> bf16 out.
// ---------------------------------------------------------------------------
__global__ __launch_bounds__(256) void rotary_kernel(
    const float* __restrict__ x, const float* __restrict__ emb,
    bf16_t* __restrict__ y, int mode) {
    const int row = blockIdx.x, tid = threadIdx.x;
    const float* xr = x + (size_t)row * DMODEL;
#pragma unroll
    for (int it = 0; it < 4; ++it) {
        int j = tid + it * 256;
        int i = j & 63;
        float f;
        if (mode == 0) {
            int pos = row & 63;
            f = (pos == 0) ? emb[63 * 64 + i] : 0.0f;
        } else {
            int pos = row & 255;
            f = emb[(pos & 127) * 64 + i];
        }
        float t = xr[j];
        float partner = (i < 32) ? -xr[j + 32] : xr[j - 32];
        y[(size_t)row * DMODEL + j] = (bf16_t)(t * __cosf(f) + partner * __sinf(f));
    }
}

// ---------------------------------------------------------------------------
// xq = shifted hs rows (token = c*64 + r + 63, zero past end) as bf16
// ---------------------------------------------------------------------------
__global__ __launch_bounds__(128) void build_xq_kernel(
    const bf16_t* __restrict__ hs, bf16_t* __restrict__ xq) {
    const int row = blockIdx.x;       // 0..4095
    const int cg = row >> 6;          // global chunk
    const int b = cg >> 4, c = cg & 15, r = row & 63;
    const int token = c * 64 + r + 63;
    u32x4* dst = reinterpret_cast<u32x4*>(xq + (size_t)row * DMODEL);
    if (token < NTOK) {
        const u32x4* src = reinterpret_cast<const u32x4*>(
            hs + ((size_t)(b * NTOK + token)) * DMODEL);
        for (int i = threadIdx.x; i < 128; i += 128) dst[i] = src[i];
    } else {
        u32x4 z = (u32x4)0u;
        for (int i = threadIdx.x; i < 128; i += 128) dst[i] = z;
    }
}

__global__ void f32_to_bf16_kernel(const float* __restrict__ x,
                                   bf16_t* __restrict__ y, size_t n) {
    size_t i = (size_t)blockIdx.x * blockDim.x + threadIdx.x;
    size_t stride = (size_t)gridDim.x * blockDim.x;
    for (; i < n; i += stride) y[i] = (bf16_t)x[i];
}

__global__ void add_kernel(const float* __restrict__ a,
                           const float* __restrict__ b, float* __restrict__ o,
                           size_t n) {
    size_t i = (size_t)blockIdx.x * blockDim.x + threadIdx.x;
    size_t stride = (size_t)gridDim.x * blockDim.x;
    for (; i < n; i += stride) o[i] = a[i] + b[i];
}

// ---------------------------------------------------------------------------
// Host orchestration
// ---------------------------------------------------------------------------
static inline void launch_gemm(const bf16_t* A, const bf16_t* B,
                               const float* bias, float* Cf, bf16_t* Cb, int M,
                               int N, int K, float alpha, int gelu,
                               hipStream_t stream) {
    dim3 grid(N / 128, M / 128);
    gemm_bf16_kernel<<<grid, 256, 0, stream>>>(A, B, bias, Cf, Cb, M, N, K,
                                               alpha, gelu);
}

extern "C" void kernel_launch(void* const* d_in, const int* in_sizes, int n_in,
                              void* d_out, int out_size, void* d_ws,
                              size_t ws_size, hipStream_t stream) {
    // Inputs (setup_inputs order)
    const float* hidden = (const float*)d_in[0];
    const float* retrieval = (const float*)d_in[1];
    const float* q_pos_emb = (const float*)d_in[2];
    const float* k_pos_emb = (const float*)d_in[3];
    const float* ln1_g = (const float*)d_in[4];
    const float* ln1_b = (const float*)d_in[5];
    const float* Wq_s = (const float*)d_in[6];
    const float* Wk_s = (const float*)d_in[7];
    const float* Wv_s = (const float*)d_in[8];
    const float* Wo_s = (const float*)d_in[9];
    const float* bo_s = (const float*)d_in[10];
    const float* ln2_g = (const float*)d_in[11];
    const float* ln2_b = (const float*)d_in[12];
    const float* Wq_c = (const float*)d_in[13];
    const float* Wk_c = (const float*)d_in[14];
    const float* Wv_c = (const float*)d_in[15];
    const float* Wo_c = (const float*)d_in[16];
    const float* bo_c = (const float*)d_in[17];
    const float* null_k = (const float*)d_in[18];
    const float* null_v = (const float*)d_in[19];
    const float* post_w = (const float*)d_in[20];
    const float* W1 = (const float*)d_in[21];
    const float* b1 = (const float*)d_in[22];
    const float* W2 = (const float*)d_in[23];
    const float* b2 = (const float*)d_in[24];
    float* out = (float*)d_out;

    // Bump allocator over d_ws
    char* ws = (char*)d_ws;
    size_t off = 0;
    auto alloc = [&](size_t bytes) -> void* {
        void* p = ws + off;
        off += (bytes + 255) & ~(size_t)255;
        return p;
    };
    const size_t DD = (size_t)DMODEL * DMODEL;          // 1M
    const size_t TOKD = (size_t)ROWS * DMODEL;          // 4M
    const size_t CTXD = (size_t)KROWS * DMODEL;         // 16M
    const size_t FFD = (size_t)DMODEL * DFF;            // 4M

    bf16_t* wq_s_b = (bf16_t*)alloc(DD * 2);
    bf16_t* wk_s_b = (bf16_t*)alloc(DD * 2);
    bf16_t* wv_s_b = (bf16_t*)alloc(DD * 2);
    bf16_t* wo_s_b = (bf16_t*)alloc(DD * 2);
    bf16_t* wq_c_b = (bf16_t*)alloc(DD * 2);
    bf16_t* wk_c_b = (bf16_t*)alloc(DD * 2);
    bf16_t* wv_c_b = (bf16_t*)alloc(DD * 2);
    bf16_t* wo_c_b = (bf16_t*)alloc(DD * 2);
    bf16_t* w1_b = (bf16_t*)alloc(FFD * 2);
    bf16_t* w2_b = (bf16_t*)alloc(FFD * 2);
    bf16_t* x_b = (bf16_t*)alloc(TOKD * 2);
    bf16_t* q_s = (bf16_t*)alloc(TOKD * 2);
    bf16_t* k_s = (bf16_t*)alloc(TOKD * 2);
    bf16_t* v_s = (bf16_t*)alloc(TOKD * 2);
    bf16_t* attn_s = (bf16_t*)alloc(TOKD * 2);
    float* oproj_f = (float*)alloc(TOKD * 4);
    float* hs_f = (float*)alloc(TOKD * 4);      // residual (kept)
    bf16_t* hs_b = (bf16_t*)alloc(TOKD * 2);
    bf16_t* xq_b = (bf16_t*)alloc(TOKD * 2);
    bf16_t* ctx_b = (bf16_t*)alloc(CTXD * 2);
    float* qc_f = (float*)alloc(TOKD * 4);
    bf16_t* qc_b = (bf16_t*)alloc(TOKD * 2);
    float* kc_f = (float*)alloc(CTXD * 4);
    bf16_t* kc_b = (bf16_t*)alloc(CTXD * 2);
    bf16_t* vc_b = (bf16_t*)alloc(CTXD * 2);
    bf16_t* attn_c = (bf16_t*)alloc(TOKD * 2);
    float* cca_f = (float*)alloc(TOKD * 4);
    bf16_t* hs2_b = (bf16_t*)alloc(TOKD * 2);
    bf16_t* ff1_b = (bf16_t*)alloc((size_t)ROWS * DFF * 2);
    float* ff2_f = (float*)alloc(TOKD * 4);
    (void)ws_size; (void)n_in; (void)in_sizes; (void)out_size;

    const float scale = 0.125f;  // DIM_HEAD^-0.5

    // weights -> bf16
    f32_to_bf16_kernel<<<1024, 256, 0, stream>>>(Wq_s, wq_s_b, DD);
    f32_to_bf16_kernel<<<1024, 256, 0, stream>>>(Wk_s, wk_s_b, DD);
    f32_to_bf16_kernel<<<1024, 256, 0, stream>>>(Wv_s, wv_s_b, DD);
    f32_to_bf16_kernel<<<1024, 256, 0, stream>>>(Wo_s, wo_s_b, DD);
    f32_to_bf16_kernel<<<1024, 256, 0, stream>>>(Wq_c, wq_c_b, DD);
    f32_to_bf16_kernel<<<1024, 256, 0, stream>>>(Wk_c, wk_c_b, DD);
    f32_to_bf16_kernel<<<1024, 256, 0, stream>>>(Wv_c, wv_c_b, DD);
    f32_to_bf16_kernel<<<1024, 256, 0, stream>>>(Wo_c, wo_c_b, DD);
    f32_to_bf16_kernel<<<2048, 256, 0, stream>>>(W1, w1_b, FFD);
    f32_to_bf16_kernel<<<2048, 256, 0, stream>>>(W2, w2_b, FFD);
    f32_to_bf16_kernel<<<4096, 256, 0, stream>>>(retrieval, ctx_b, CTXD);

    // ---- self attention ----
    ln_kernel<<<ROWS, 256, 0, stream>>>(hidden, nullptr, ln1_g, ln1_b, nullptr,
                                        x_b);
    launch_gemm(x_b, wq_s_b, nullptr, nullptr, q_s, ROWS, DMODEL, DMODEL, scale,
                0, stream);
    launch_gemm(x_b, wk_s_b, nullptr, nullptr, k_s, ROWS, DMODEL, DMODEL, 1.0f,
                0, stream);
    launch_gemm(x_b, wv_s_b, nullptr, nullptr, v_s, ROWS, DMODEL, DMODEL, 1.0f,
                0, stream);
    attn_self_kernel<<<dim3(16, 64), 128, 0, stream>>>(q_s, k_s, v_s, attn_s);
    launch_gemm(attn_s, wo_s_b, bo_s, oproj_f, nullptr, ROWS, DMODEL, DMODEL,
                1.0f, 0, stream);
    ln_kernel<<<ROWS, 256, 0, stream>>>(oproj_f, hidden, ln2_g, ln2_b, hs_f,
                                        hs_b);

    // ---- chunked cross attention ----
    build_xq_kernel<<<ROWS, 128, 0, stream>>>(hs_b, xq_b);
    launch_gemm(xq_b, wq_c_b, nullptr, qc_f, nullptr, ROWS, DMODEL, DMODEL,
                scale, 0, stream);
    launch_gemm(ctx_b, wk_c_b, nullptr, kc_f, nullptr, KROWS, DMODEL, DMODEL,
                1.0f, 0, stream);
    launch_gemm(ctx_b, wv_c_b, nullptr, nullptr, vc_b, KROWS, DMODEL, DMODEL,
                1.0f, 0, stream);
    rotary_kernel<<<ROWS, 256, 0, stream>>>(qc_f, q_pos_emb, qc_b, 0);
    rotary_kernel<<<KROWS, 256, 0, stream>>>(kc_f, k_pos_emb, kc_b, 1);
    attn_cross_kernel<<<dim3(NCHUNK, HEADS), 128, 0, stream>>>(
        qc_b, kc_b, vc_b, null_k, null_v, attn_c);
    launch_gemm(attn_c, wo_c_b, bo_c, cca_f, nullptr, ROWS, DMODEL, DMODEL,
                1.0f, 0, stream);
    t5norm_kernel<<<ROWS, 256, 0, stream>>>(hs_f, cca_f, post_w, hs2_b);

    // ---- FFN ----
    launch_gemm(hs2_b, w1_b, b1, nullptr, ff1_b, ROWS, DFF, DMODEL, 1.0f, 1,
                stream);
    launch_gemm(ff1_b, w2_b, b2, ff2_f, nullptr, ROWS, DMODEL, DFF, 1.0f, 0,
                stream);
    add_kernel<<<2048, 256, 0, stream>>>(hs_f, ff2_f, out, TOKD);
}
